// SDMAttention_78529182040515
// MI455X (gfx1250) — matrix operand order repaired and checked by reading, not compiled, not run on hardware
//
#include <hip/hip_runtime.h>
#include <hip/hip_bf16.h>
#include <math.h>

// ---------------------------------------------------------------------------
// SDMAttention for MI455X (gfx1250).
// After dead-code elimination the reference reduces to B*N scalars c[b][s]
// broadcast over (i, e), so the kernel is bandwidth-bound streaming:
//   K1: per (b,i) -> R = sum(u0 row), sm = softmax_E(w0[b,i,23,:]),
//       dh[e] = sum_h (y-x), and two 512-length dots done with
//       v_wmma_f32_16x16x32_f16 via the trace trick. values slabs are staged
//       into LDS with CDNA5 async global->LDS copies.
//   K2: per-batch sequential recurrence over s (1 wave per batch).
//   K3: broadcast + mask fill of the 151MB output (float4 stores).
// ---------------------------------------------------------------------------

typedef __attribute__((ext_vector_type(16))) _Float16 v16h;
typedef __attribute__((ext_vector_type(8)))  float    v8f;

#define NN   24
#define EE   512
#define HH   8
#define VB_STRIDE (48 * HH * EE)       // 196608, per-batch stride of values
#define ROW       (HH * EE)            // 4096, one (l) slab of values / u0 row
#define CNT_INV   (1.0f / 98304.0f)    // 1 / (N*H*E)

__device__ __forceinline__ float wredsum(float v) {
#pragma unroll
  for (int o = 16; o > 0; o >>= 1) v += __shfl_xor(v, o, 32);
  return v;
}
__device__ __forceinline__ float wredmax(float v) {
#pragma unroll
  for (int o = 16; o > 0; o >>= 1) v = fmaxf(v, __shfl_xor(v, o, 32));
  return v;
}

// ---------------------------------------------------------------------------
// Kernel 1: one block per (b, i).  256 threads = 8 waves.
// Outputs (workspace): Rws[b*24+i], Apart[b*24+i] (= sum_e sm*dh),
//                      Dpart[b*24+i] (= sum_e dh).
// ---------------------------------------------------------------------------
__global__ __launch_bounds__(256) void sdm_k1_reduce(
    const float* __restrict__ values, const float* __restrict__ u0,
    const float* __restrict__ w0, float* __restrict__ Rws,
    float* __restrict__ Apart, float* __restrict__ Dpart) {
  __shared__ __align__(16) float s_x[ROW];   // values[b, i, :, :]
  __shared__ __align__(16) float s_y[ROW];   // values[b, 24+i, :, :]
  __shared__ float s_dh[EE];                 // sum over h of (y - x)
  __shared__ float s_sm[EE];                 // exp(w0 - max) (unnormalized)
  __shared__ float s_red[16];

  const int t  = threadIdx.x;
  const int bi = blockIdx.x;
  const int b  = bi / NN;
  const int i  = bi % NN;

  const float* gx    = values + (size_t)b * VB_STRIDE + (size_t)i * ROW;
  const float* gy    = values + (size_t)b * VB_STRIDE + (size_t)(NN + i) * ROW;
  const float* w0row = w0 + (size_t)b * (NN * NN * EE) + (size_t)i * (NN * EE) +
                       (size_t)(NN - 1) * EE;
  const float* u0row = u0 + (size_t)b * (NN * ROW) + (size_t)i * ROW;

  __builtin_prefetch(w0row, 0, 0);  // global_prefetch_b8

  // ---- CDNA5 async global->LDS staging of both 16KB value slabs ----------
  const unsigned int lx = (unsigned int)(uintptr_t)(&s_x[0]);
  const unsigned int ly = (unsigned int)(uintptr_t)(&s_y[0]);
#pragma unroll
  for (int k = 0; k < 4; ++k) {
    const int f = (k * 256 + t) * 4;  // float index, 16B per thread per op
    asm volatile("global_load_async_to_lds_b128 %0, %1, off"
                 :: "v"(lx + (unsigned int)(f * 4)), "v"(gx + f) : "memory");
    asm volatile("global_load_async_to_lds_b128 %0, %1, off"
                 :: "v"(ly + (unsigned int)(f * 4)), "v"(gy + f) : "memory");
  }
  asm volatile("s_wait_asynccnt 0" ::: "memory");
  __syncthreads();

  // ---- dh[e] = sum_h (y - x) ---------------------------------------------
  for (int e = t; e < EE; e += 256) {
    float acc = 0.f;
#pragma unroll
    for (int h = 0; h < HH; ++h) acc += s_y[h * EE + e] - s_x[h * EE + e];
    s_dh[e] = acc;
  }

  // ---- softmax numerators of w0[b, i, 23, :] over E ----------------------
  float wv[2];
  float lm = -3.0e38f;
#pragma unroll
  for (int k = 0; k < 2; ++k) {
    wv[k] = w0row[t + 256 * k];
    lm = fmaxf(lm, wv[k]);
  }
  lm = wredmax(lm);
  if ((t & 31) == 0) s_red[t >> 5] = lm;
  __syncthreads();  // also publishes s_dh
  float mx = s_red[0];
#pragma unroll
  for (int k = 1; k < 8; ++k) mx = fmaxf(mx, s_red[k]);

  float ls = 0.f;
#pragma unroll
  for (int k = 0; k < 2; ++k) {
    const float ex = __expf(wv[k] - mx);
    s_sm[t + 256 * k] = ex;
    ls += ex;
  }
  ls = wredsum(ls);
  if ((t & 31) == 0) s_red[8 + (t >> 5)] = ls;
  __syncthreads();
  float tot = 0.f;
#pragma unroll
  for (int k = 0; k < 8; ++k) tot += s_red[8 + k];
  const float inv = 1.0f / tot;  // softmax normalizer, applied to dot result

  // ---- R = sum of u0 row (4096 floats) -----------------------------------
  float racc = 0.f;
#pragma unroll
  for (int k = 0; k < 16; ++k) racc += u0row[t + 256 * k];
  racc = wredsum(racc);
  if ((t & 31) == 0) s_red[t >> 5] = racc;
  __syncthreads();
  if (t == 0) {
    float r = 0.f;
#pragma unroll
    for (int k = 0; k < 8; ++k) r += s_red[k];
    Rws[bi] = r;
  }

  // ---- wave 0: trace-trick dots via v_wmma_f32_16x16x32_f16 --------------
  // A[m,k] = sm[32m+k], B[k,n] = dh[32n+k]  =>  trace(AxB) = dot(sm, dh).
  // A = ones                               =>  trace       = sum(dh).
  if (t < 32) {
    const int lane = t;
    const int M  = lane & 15;
    const int hi = lane >> 4;

    v16h a, ones, bm;
#pragma unroll
    for (int j = 0; j < 8; ++j) {
      a[j]     = (_Float16)s_sm[32 * M + 16 * hi + j];          // K = 16*hi + j
      a[8 + j] = (_Float16)s_sm[32 * M + 16 * hi + 8 + j];      // K = 16*hi+8+j
    }
#pragma unroll
    for (int j = 0; j < 16; ++j) {
      bm[j]   = (_Float16)s_dh[32 * M + 16 * hi + j];           // K = 16*hi + j
      ones[j] = (_Float16)1.0f;
    }

    v8f c0 = {};
    v8f c1 = {};
    c0 = __builtin_amdgcn_wmma_f32_16x16x32_f16(false, a, false, bm,
                                                (short)0, c0, false, false);
    c1 = __builtin_amdgcn_wmma_f32_16x16x32_f16(false, ones, false, bm,
                                                (short)0, c1, false, false);

    // diagonal element (m,m): m<8 -> lane m, vgpr m ; m>=8 -> lane m+16, vgpr m-8
    const int m = (lane < 8) ? lane : ((lane >= 24) ? (lane - 24) : -1);
    float d0 = 0.f, d1 = 0.f;
#pragma unroll
    for (int k = 0; k < 8; ++k) {
      d0 += (m == k) ? c0[k] : 0.f;
      d1 += (m == k) ? c1[k] : 0.f;
    }
    d0 = wredsum(d0);
    d1 = wredsum(d1);
    if (lane == 0) {
      Apart[bi] = d0 * inv;  // sum_e softmax(w0)[e] * dh[e]
      Dpart[bi] = d1;        // sum_e dh[e]
    }
  }
}

// ---------------------------------------------------------------------------
// Kernel 2: one wave per batch; sequential recurrence over s = 0..23.
// ---------------------------------------------------------------------------
__global__ __launch_bounds__(32) void sdm_k2_recur(
    const float* __restrict__ values, const float* __restrict__ Rws,
    const float* __restrict__ Apart, const float* __restrict__ Dpart,
    float* __restrict__ Cws) {
  const int b    = blockIdx.x;
  const int lane = threadIdx.x;

  float Rl = 0.f, Al = 0.f, Dl = 0.f;
  if (lane < NN) {
    Rl = Rws[b * NN + lane];
    Al = Apart[b * NN + lane];
    Dl = Dpart[b * NN + lane];
  }
  const float U0  = wredsum(Rl);  // sum over all of u0[b]
  const float A   = wredsum(Al);  // rowsum at s = 0
  const float Dt  = wredsum(Dl);  // sum over all of d[b]

  // dcol[v][s] = d[b, v, 0, s] = y - x at head 0, first 24 e-channels
  float dv[NN];
  if (lane < NN) {
    const float* px = values + (size_t)b * VB_STRIDE + (size_t)lane * ROW;
    const float* py = px + (size_t)NN * ROW;
#pragma unroll
    for (int s = 0; s < NN; ++s) dv[s] = py[s] - px[s];
  } else {
#pragma unroll
    for (int s = 0; s < NN; ++s) dv[s] = 0.f;
  }

  float S = U0;
  float cprev = 0.f;
#pragma unroll
  for (int s = 0; s < NN; ++s) {
    const float mean = S * CNT_INV;
    const float lamb = mean * mean;
    const float tv   = (lane < NN) ? (-lamb * dv[s]) : -3.0e38f;
    const float mx   = wredmax(tv);
    const float ex   = (lane < NN) ? __expf(tv - mx) : 0.f;
    const float ssum = wredsum(ex);
    const float e23  = __shfl(ex, NN - 1, 32);
    const float c    = e23 / ssum;

    const float rowsum = (s == 0) ? A : (cprev * Dt);
    const float Rs     = __shfl(Rl, s, 32);
    S = S - Rs + rowsum;
    cprev = c;
    if (lane == 0) Cws[b * NN + s] = c;
  }
}

// ---------------------------------------------------------------------------
// Kernel 3: fill output (B, 48, 24, 512): l<24 -> 1e-9, else c[b][s];
// masked positions -> -inf.  4 elements (and 4 mask bytes) per thread.
// ---------------------------------------------------------------------------
__global__ __launch_bounds__(256) void sdm_k3_fill(
    const unsigned char* __restrict__ mask, const float* __restrict__ Cws,
    float* __restrict__ out, int n4) {
  const int g = blockIdx.x * 256 + threadIdx.x;
  if (g >= n4) return;

  int r = g >> 7;               // drop e4 (512/4 = 128 groups per (b,l,s))
  const int s = r % NN;  r /= NN;
  const int l = r % 48;
  const int b = r / 48;

  const float base = (l < NN) ? 1e-9f : Cws[b * NN + s];
  const unsigned int m = *(const unsigned int*)(mask + (size_t)g * 4);
  const float ninf = -__builtin_inff();

  float4 o;
  o.x = (m & 0x000000ffu) ? ninf : base;
  o.y = (m & 0x0000ff00u) ? ninf : base;
  o.z = (m & 0x00ff0000u) ? ninf : base;
  o.w = (m & 0xff000000u) ? ninf : base;
  *(float4*)(out + (size_t)g * 4) = o;
}

// ---------------------------------------------------------------------------
// Host-side launcher.
// Inputs: 0 queries, 1 keys, 2 values, 3 v0, 4 u0, 5 w0, 6 attn_mask (bool).
// ---------------------------------------------------------------------------
extern "C" void kernel_launch(void* const* d_in, const int* in_sizes, int n_in,
                              void* d_out, int out_size, void* d_ws,
                              size_t ws_size, hipStream_t stream) {
  const float* values = (const float*)d_in[2];
  const float* u0     = (const float*)d_in[4];
  const float* w0     = (const float*)d_in[5];
  const unsigned char* mask = (const unsigned char*)d_in[6];

  const int B = in_sizes[2] / VB_STRIDE;  // 64

  float* ws    = (float*)d_ws;
  float* Rws   = ws;
  float* Apart = ws + (size_t)B * NN;
  float* Dpart = ws + (size_t)2 * B * NN;
  float* Cws   = ws + (size_t)3 * B * NN;

  sdm_k1_reduce<<<B * NN, 256, 0, stream>>>(values, u0, w0, Rws, Apart, Dpart);
  sdm_k2_recur<<<B, 32, 0, stream>>>(values, Rws, Apart, Dpart, Cws);

  const int n4 = out_size / 4;
  sdm_k3_fill<<<(n4 + 255) / 256, 256, 0, stream>>>(mask, Cws, (float*)d_out,
                                                    n4);
}
